// MultiHeadCrossAttention_15796889715179
// MI455X (gfx1250) — matrix operand order repaired and checked by reading, not compiled, not run on hardware
//
#include <hip/hip_runtime.h>
#include <cstddef>
#include <cstdint>

// ---------------------------------------------------------------------------
// MultiHeadCrossAttention on gfx1250 (MI455X): bf16 WMMA everywhere.
// B=4, SQ=1024, SK=2048, EMBED=1024, H=16, Dh=64, scale = 1/8.
// ---------------------------------------------------------------------------

typedef __bf16 bf16_t;
typedef __attribute__((ext_vector_type(16))) __bf16 bfx16;
typedef __attribute__((ext_vector_type(8)))  __bf16 bfx8;
typedef __attribute__((ext_vector_type(8)))  float  fx8;
typedef __attribute__((ext_vector_type(4)))  float  fx4;

#define EMBED 1024
#define HEADS 16
#define HDIM  64
#define SEQ_Q 1024
#define SEQ_K 2048
#define BATCH 4

#define PREF(p) __builtin_prefetch((const void*)(p), 0, 1)

// ---------------- helpers -------------------------------------------------

__device__ __forceinline__ bf16_t f2bf(float f) { return (bf16_t)f; }

__device__ __forceinline__ fx8 zero8() {
  fx8 z;
#pragma unroll
  for (int i = 0; i < 8; ++i) z[i] = 0.0f;
  return z;
}

__device__ __forceinline__ fx8 wmma_bf16(bfx16 a, bfx16 b, fx8 c) {
  return __builtin_amdgcn_wmma_f32_16x16x32_bf16(
      /*neg_a=*/false, a, /*neg_b=*/false, b,
      /*c_mod=*/(short)0, c, /*reuse_a=*/false, /*reuse_b=*/false);
}

// A-fragment (16x32 bf16): lane holds row M=lane%16; element e maps to
// K = 16*half + e (contiguous).
__device__ __forceinline__ bfx16 afrag_load(const bf16_t* row, int half) {
  bfx8 c0 = *(const bfx8*)(row + 16 * half);
  bfx8 c1 = *(const bfx8*)(row + 16 * half + 8);
  bfx16 a;
#pragma unroll
  for (int i = 0; i < 8; ++i) { a[i] = c0[i]; a[8 + i] = c1[i]; }
  return a;
}
__device__ __forceinline__ bfx16 afrag_load(const float* row, int half) {
  fx8 c0 = *(const fx8*)(row + 16 * half);
  fx8 c1 = *(const fx8*)(row + 16 * half + 8);
  bfx16 a;
#pragma unroll
  for (int i = 0; i < 8; ++i) { a[i] = f2bf(c0[i]); a[8 + i] = f2bf(c1[i]); }
  return a;
}

// B-fragment (32x16 bf16): lane holds col N=lane%16; element e maps to
// K = 16*half + e -> 16 contiguous elements starting at K=16*half.
__device__ __forceinline__ bfx16 bfrag_load(const bf16_t* p16) {
  bfx8 c0 = *(const bfx8*)(p16);
  bfx8 c1 = *(const bfx8*)(p16 + 8);
  bfx16 b;
#pragma unroll
  for (int i = 0; i < 8; ++i) { b[i] = c0[i]; b[8 + i] = c1[i]; }
  return b;
}

__device__ __forceinline__ void store_out(float*  p, float v) { *p = v; }
__device__ __forceinline__ void store_out(bf16_t* p, float v) { *p = f2bf(v); }

// ---------------- bulk f32 -> bf16 conversion (weights, once per launch) ---

__global__ void __launch_bounds__(256)
cvt_kernel(const float* __restrict__ in, bf16_t* __restrict__ out) {
  const size_t i = ((size_t)blockIdx.x * 256 + threadIdx.x) * 8;
  fx8 v = *(const fx8*)(in + i);
  bfx8 o;
#pragma unroll
  for (int j = 0; j < 8; ++j) o[j] = f2bf(v[j]);
  *(bfx8*)(out + i) = o;
}

// ---------------- projection GEMM: out[M,1024] = x[M,1024] @ Wb^T + bias ---
// Wb is pre-converted bf16. grid = (1024/128, M/64), block = 256 (8 waves).

template <typename InT, typename OutT>
__global__ void __launch_bounds__(256)
proj_kernel(const InT* __restrict__ x, const bf16_t* __restrict__ Wb,
            const float* __restrict__ bias, OutT* __restrict__ out) {
  const int wave = threadIdx.x >> 5;
  const int lane = threadIdx.x & 31;
  const int half = lane >> 4;
  const int lm   = lane & 15;
  const int wm   = wave >> 2;            // 0..1
  const int wn   = wave & 3;             // 0..3
  const int m0   = blockIdx.y * 64 + wm * 32;
  const int n0   = blockIdx.x * 128 + wn * 32;

  fx8 acc[2][2];
#pragma unroll
  for (int i = 0; i < 2; ++i)
#pragma unroll
    for (int j = 0; j < 2; ++j) acc[i][j] = zero8();

  const InT*    xr0 = x  + (size_t)(m0 + lm) * EMBED;
  const InT*    xr1 = x  + (size_t)(m0 + 16 + lm) * EMBED;
  const bf16_t* wr0 = Wb + (size_t)(n0 + lm) * EMBED;
  const bf16_t* wr1 = Wb + (size_t)(n0 + 16 + lm) * EMBED;

  for (int k0 = 0; k0 < EMBED; k0 += 32) {
    // speculative prefetch of next k-step (silently dropped if OOB)
    PREF(xr0 + k0 + 32); PREF(xr1 + k0 + 32);
    PREF(wr0 + k0 + 32); PREF(wr1 + k0 + 32);
    bfx16 a0 = afrag_load(xr0 + k0, half);
    bfx16 a1 = afrag_load(xr1 + k0, half);
    bfx16 b0 = bfrag_load(wr0 + k0 + 16 * half);
    bfx16 b1 = bfrag_load(wr1 + k0 + 16 * half);
    acc[0][0] = wmma_bf16(a0, b0, acc[0][0]);
    acc[0][1] = wmma_bf16(a0, b1, acc[0][1]);
    acc[1][0] = wmma_bf16(a1, b0, acc[1][0]);
    acc[1][1] = wmma_bf16(a1, b1, acc[1][1]);
  }

#pragma unroll
  for (int nt = 0; nt < 2; ++nt) {
    const int n = n0 + nt * 16 + lm;
    const float bv = bias[n];
#pragma unroll
    for (int mt = 0; mt < 2; ++mt) {
#pragma unroll
      for (int r = 0; r < 8; ++r) {
        const int m = m0 + mt * 16 + r + 8 * half;   // C/D layout
        store_out(out + (size_t)m * EMBED + n, acc[mt][nt][r] + bv);
      }
    }
  }
}

// ---------------- attention kernel ----------------------------------------
// One block per (b, h, 16-query tile). LDS: scores 16x2052 f32 (padded),
// V stage 2x32x80 bf16, partial-C 8x32x8 f32.

#define SC_STRIDE 2052
#define VT_STRIDE 80
#define SMEM_SC   (16 * SC_STRIDE * 4)        // 131328 B
#define SMEM_VT   (2 * 32 * VT_STRIDE * 2)    //  10240 B
#define SMEM_PART (8 * 32 * 8 * 4)            //   8192 B
#define SMEM_TOTAL (SMEM_SC + SMEM_VT + SMEM_PART)

__global__ void __launch_bounds__(256)
attn_kernel(const bf16_t* __restrict__ qb, const bf16_t* __restrict__ kb,
            const bf16_t* __restrict__ vb, bf16_t* __restrict__ attb,
            float* __restrict__ mean_out) {
  extern __shared__ char smem[];
  float*  sc   = (float*)smem;
  bf16_t* vt   = (bf16_t*)(smem + SMEM_SC);
  float*  part = (float*)(smem + SMEM_SC + SMEM_VT);

  const int bi = blockIdx.x;
  const int qt = bi & 63;
  const int h  = (bi >> 6) & 15;
  const int b  = bi >> 10;
  const int q0 = qt * 16;

  const int wave = threadIdx.x >> 5;
  const int lane = threadIdx.x & 31;
  const int half = lane >> 4;
  const int lm   = lane & 15;

  // ---- phase 1: scores = (q*scale) @ k^T,  M=16, N=2048, K(head dim)=64 --
  const bf16_t* qrow = qb + ((size_t)(b * SEQ_Q + q0 + lm)) * EMBED + h * HDIM;
  bfx16 aq0 = afrag_load(qrow, half);        // dims 0..31
  bfx16 aq1 = afrag_load(qrow + 32, half);   // dims 32..63
  // fold softmax scale 1/8 into Q fragments: power of two -> exact in bf16,
  // and removes the dependent multiply from the WMMA -> LDS-store chain.
#pragma unroll
  for (int e = 0; e < 16; ++e) {
    aq0[e] = f2bf(0.125f * (float)aq0[e]);
    aq1[e] = f2bf(0.125f * (float)aq1[e]);
  }

  // two n-tiles in flight -> two independent WMMA accumulator chains
  for (int nt = 0; nt < 16; nt += 2) {
    const int sg0 = wave * 256 + nt * 16 + lm;       // key index 0..2047
    const int sg1 = sg0 + 16;
    const bf16_t* kr0 = kb + ((size_t)(b * SEQ_K + sg0)) * EMBED + h * HDIM;
    const bf16_t* kr1 = kb + ((size_t)(b * SEQ_K + sg1)) * EMBED + h * HDIM;
    PREF(kr0 + 32 * EMBED);                          // keys two tiles ahead
    PREF(kr1 + 32 * EMBED);
    bfx16 b00 = bfrag_load(kr0 + 16 * half);
    bfx16 b01 = bfrag_load(kr0 + 32 + 16 * half);
    bfx16 b10 = bfrag_load(kr1 + 16 * half);
    bfx16 b11 = bfrag_load(kr1 + 32 + 16 * half);
    fx8 acc0 = zero8();
    fx8 acc1 = zero8();
    acc0 = wmma_bf16(aq0, b00, acc0);
    acc1 = wmma_bf16(aq0, b10, acc1);
    acc0 = wmma_bf16(aq1, b01, acc0);
    acc1 = wmma_bf16(aq1, b11, acc1);
#pragma unroll
    for (int r = 0; r < 8; ++r) {
      sc[(r + 8 * half) * SC_STRIDE + sg0] = acc0[r];
      sc[(r + 8 * half) * SC_STRIDE + sg1] = acc1[r];
    }
  }
  __syncthreads();

  // ---- phase 2: softmax per row; accumulate mean over heads via atomics --
#pragma unroll
  for (int rr = 0; rr < 2; ++rr) {
    const int row = wave * 2 + rr;
    float* srow = sc + row * SC_STRIDE;
    float mx = -3.0e38f;
    for (int i = lane; i < SEQ_K; i += 32) mx = fmaxf(mx, srow[i]);
#pragma unroll
    for (int off = 16; off > 0; off >>= 1) mx = fmaxf(mx, __shfl_xor(mx, off));
    float sum = 0.0f;
    for (int i = lane; i < SEQ_K; i += 32) {
      float e = __expf(srow[i] - mx);
      srow[i] = e;
      sum += e;
    }
#pragma unroll
    for (int off = 16; off > 0; off >>= 1) sum += __shfl_xor(sum, off);
    const float inv = 1.0f / sum;
    float* mrow = mean_out + ((size_t)(b * SEQ_Q + q0 + row)) * SEQ_K;
    for (int i = lane; i < SEQ_K; i += 32) {
      float p = srow[i] * inv;
      srow[i] = p;
      atomicAdd(mrow + i, p * (1.0f / 16.0f));
    }
  }

  // ---- phase 3: attended = P @ V,  M=16, N=64, K(keys)=2048 --------------
  // wave -> C tile nt3 = wave&3 (16 cols of head dim), K-half g = wave>>2.
  const int nt3 = wave & 3;
  const int g   = wave >> 2;
  fx8 acc = zero8();

  const int tg  = threadIdx.x >> 7;   // which K-half this thread stages
  const int tj  = threadIdx.x & 127;
  const int tkl = tj >> 2;            // key 0..31 within the staged tile
  const int tcb = (tj & 3) << 4;      // dim chunk {0,16,32,48}

  for (int i = 0; i < 32; ++i) {
    __syncthreads();
    {   // stage 2 tiles of V (32 keys x 64 dims each), coalesced 32B loads
      const int key = tg * 1024 + i * 32 + tkl;
      const bf16_t* src = vb + ((size_t)(b * SEQ_K + key)) * EMBED + h * HDIM + tcb;
      PREF(src + 32 * EMBED);         // next iteration's V rows
      bfx8 d0 = *(const bfx8*)src;
      bfx8 d1 = *(const bfx8*)(src + 8);
      bf16_t* dst = vt + (tg * 32 + tkl) * VT_STRIDE + tcb;
      *(bfx8*)dst       = d0;
      *(bfx8*)(dst + 8) = d1;
    }
    __syncthreads();

    // A fragment from normalized probs in LDS (f32 -> bf16, native cvt)
    const float* prow = sc + lm * SC_STRIDE + g * 1024 + i * 32;
    fx4 c0 = *(const fx4*)(prow + 8 * half);
    fx4 c1 = *(const fx4*)(prow + 8 * half + 4);
    fx4 c2 = *(const fx4*)(prow + 16 + 8 * half);
    fx4 c3 = *(const fx4*)(prow + 16 + 8 * half + 4);
    bfx16 af;
#pragma unroll
    for (int e = 0; e < 4; ++e) {
      af[e]      = f2bf(c0[e]);
      af[4 + e]  = f2bf(c1[e]);
      af[8 + e]  = f2bf(c2[e]);
      af[12 + e] = f2bf(c3[e]);
    }
    // B fragment: B(K=key_local, N=dim) gathered from staged tile
    bfx16 bf_;
#pragma unroll
    for (int e = 0; e < 16; ++e)
      bf_[e] = vt[(g * 32 + 8 * half + (e & 7) + ((e >> 3) << 4)) * VT_STRIDE + nt3 * 16 + lm];

    acc = wmma_bf16(af, bf_, acc);
  }

  __syncthreads();
  {   // combine the two K-halves through LDS
    float* pme = part + ((size_t)wave * 32 + lane) * 8;
#pragma unroll
    for (int r = 0; r < 8; ++r) pme[r] = acc[r];
  }
  __syncthreads();
  if (wave < 4) {
    const float* pa = part + ((size_t)wave * 32 + lane) * 8;
    const float* pb = part + ((size_t)(wave + 4) * 32 + lane) * 8;
    bf16_t* obase = attb + ((size_t)(b * SEQ_Q + q0)) * EMBED + h * HDIM +
                    wave * 16 + lm;
#pragma unroll
    for (int r = 0; r < 8; ++r)
      obase[(size_t)(r + 8 * half) * EMBED] = f2bf(pa[r] + pb[r]);
  }
}

// ---------------- misc ----------------------------------------------------

__global__ void __launch_bounds__(256)
zero_kernel(float* __restrict__ p, int n) {
  int i = blockIdx.x * 256 + threadIdx.x;
  if (i < n) p[i] = 0.0f;
}

// ---------------- host launcher -------------------------------------------

extern "C" void kernel_launch(void* const* d_in, const int* in_sizes, int n_in,
                              void* d_out, int out_size, void* d_ws,
                              size_t ws_size, hipStream_t stream) {
  (void)in_sizes; (void)n_in; (void)out_size; (void)ws_size;

  const float* query = (const float*)d_in[0];
  const float* key   = (const float*)d_in[1];
  const float* value = (const float*)d_in[2];
  const float* Wq    = (const float*)d_in[3];
  const float* bq    = (const float*)d_in[4];
  const float* Wk    = (const float*)d_in[5];
  const float* bk    = (const float*)d_in[6];
  const float* Wv    = (const float*)d_in[7];
  const float* bv    = (const float*)d_in[8];
  const float* Wo    = (const float*)d_in[9];
  const float* bo    = (const float*)d_in[10];

  float* out  = (float*)d_out;                               // [4,1024,1024]
  float* mean = out + (size_t)BATCH * SEQ_Q * EMBED;         // [4,1024,2048]

  char* ws = (char*)d_ws;
  bf16_t* qb   = (bf16_t*)(ws);                              //  8 MB
  bf16_t* kb   = (bf16_t*)(ws + ((size_t)8  << 20));         // 16 MB
  bf16_t* vbuf = (bf16_t*)(ws + ((size_t)24 << 20));         // 16 MB
  bf16_t* attb = (bf16_t*)(ws + ((size_t)40 << 20));         //  8 MB
  bf16_t* wqb  = (bf16_t*)(ws + ((size_t)48 << 20));         //  2 MB
  bf16_t* wkb  = (bf16_t*)(ws + ((size_t)50 << 20));         //  2 MB
  bf16_t* wvb  = (bf16_t*)(ws + ((size_t)52 << 20));         //  2 MB
  bf16_t* wob  = (bf16_t*)(ws + ((size_t)54 << 20));         //  2 MB

  dim3 blk(256);

  // pre-convert weights to bf16 (once; removes per-tile conversion VALU)
  const int wblocks = (EMBED * EMBED) / (256 * 8);           // 512
  cvt_kernel<<<wblocks, blk, 0, stream>>>(Wq, wqb);
  cvt_kernel<<<wblocks, blk, 0, stream>>>(Wk, wkb);
  cvt_kernel<<<wblocks, blk, 0, stream>>>(Wv, wvb);
  cvt_kernel<<<wblocks, blk, 0, stream>>>(Wo, wob);

  // input projections -> bf16 workspace
  proj_kernel<float, bf16_t><<<dim3(8, 64),  blk, 0, stream>>>(query, wqb, bq, qb);
  proj_kernel<float, bf16_t><<<dim3(8, 128), blk, 0, stream>>>(key,   wkb, bk, kb);
  proj_kernel<float, bf16_t><<<dim3(8, 128), blk, 0, stream>>>(value, wvb, bv, vbuf);

  // zero mean_attn (atomically accumulated across heads)
  const int mean_n = BATCH * SEQ_Q * SEQ_K;                  // 8388608
  zero_kernel<<<mean_n / 256, blk, 0, stream>>>(mean, mean_n);

  // attention: (b, h, q-tile) blocks
  attn_kernel<<<BATCH * HEADS * (SEQ_Q / 16), blk, SMEM_TOTAL, stream>>>(
      qb, kb, vbuf, attb, mean);

  // output projection (bf16 in, f32 out)
  proj_kernel<bf16_t, float><<<dim3(8, 64), blk, 0, stream>>>(attb, wob, bo, out);
}
